// CDVAEDiffusion_7275674599864
// MI455X (gfx1250) — matrix operand order repaired and checked by reading, not compiled, not run on hardware
//
#include <hip/hip_runtime.h>
#include <hip/hip_bf16.h>
#include <math.h>

// ---------------------------------------------------------------------------
// CDVAE diffusion forward for MI455X (gfx1250), bf16 WMMA pipeline.
// Input flattening assumption: setup_inputs() dict insertion order, params
// pytree depth-first in insertion order (W before b in each linear).
// Workspace use: ~49 MB of d_ws.
//
// M=32 tiling: each workgroup produces a 32x256 output tile; every packed
// B fragment (32 B/lane) feeds two v_wmma_f32_16x16x32_bf16 (one per M-tile),
// halving B-operand traffic per FLOP vs M=16.
// ---------------------------------------------------------------------------

#define HDIM     256
#define NNODES   3072
#define NBATCH   96
#define EEDGES   60000
#define NSPEC    100

typedef __attribute__((ext_vector_type(16))) __bf16 v16bf;
typedef __attribute__((ext_vector_type(8)))  float  v8f;

union Frag { v16bf v; uint4 q[2]; unsigned short u[16]; };
union U4   { uint4 q; unsigned short u[8]; };

__device__ __forceinline__ unsigned short f2bf(float f) {
  unsigned int u = __float_as_uint(f);
  u += 0x7fffu + ((u >> 16) & 1u);            // round-to-nearest-even
  return (unsigned short)(u >> 16);
}
__device__ __forceinline__ float bf2f(unsigned short h) {
  return __uint_as_float(((unsigned int)h) << 16);
}
__device__ __forceinline__ float silu_f(float x) {
  return x / (1.0f + __expf(-x));
}

// A fragment from a row-major bf16 tile in LDS (16 rows, `stride` elems/row).
// Layout (ISA 7.12.2, 16-bit A 16x32): lane holds row lane%16;
// lanes 0-15: K {0..7, 16..23}; lanes 16-31: K {8..15, 24..31}.
__device__ __forceinline__ v16bf load_a_lds(const unsigned short* base, int stride,
                                            int kt, int lane) {
  const unsigned short* p = base + (lane & 15) * stride + kt * 32 + ((lane < 16) ? 0 : 8);
  Frag f;
  f.q[0] = *(const uint4*)(p);        // K half 0 (8 contiguous bf16, 16B)
  f.q[1] = *(const uint4*)(p + 16);   // K half 1
  return f.v;
}

// B fragment from pre-packed weights: 16 contiguous bf16 per lane (32B).
__device__ __forceinline__ v16bf load_b_pk(const unsigned short* wp, int kt, int nt,
                                           int NT, int lane) {
  const unsigned short* p = wp + ((((size_t)kt * NT + nt) * 32 + lane) << 4);
  Frag f;
  f.q[0] = *(const uint4*)(p);
  f.q[1] = *(const uint4*)(p + 8);
  return f.v;
}

__device__ __forceinline__ v8f wmma_bf16(v16bf a, v16bf b, v8f c) {
  return __builtin_amdgcn_wmma_f32_16x16x32_bf16(false, a, false, b, (short)0, c,
                                                 false, false);
}

// 4 waves x 4 N-tiles x 2 M-tiles = 32x256 output tile, K = KT*32.
// Each B fragment is reused by two WMMAs (M-tiles 0 and 1).
__device__ __forceinline__ void gemm_32xN(const unsigned short* a_lds, int a_stride,
                                          const unsigned short* bp, int KT,
                                          int lane, int wv, v8f acc[2][4]) {
  for (int kt = 0; kt < KT; ++kt) {
    v16bf a0 = load_a_lds(a_lds, a_stride, kt, lane);
    v16bf a1 = load_a_lds(a_lds + 16 * a_stride, a_stride, kt, lane);
#pragma unroll
    for (int j = 0; j < 4; ++j) {
      v16bf b = load_b_pk(bp, kt, wv * 4 + j, 16, lane);
      acc[0][j] = wmma_bf16(a0, b, acc[0][j]);
      acc[1][j] = wmma_bf16(a1, b, acc[1][j]);
    }
  }
}

// Epilogue: +bias, optional SiLU, store bf16 to 32x256 LDS tile.
// D layout (ISA 7.12.2): lane holds col lane%16, rows v + 8*(lane>=16).
__device__ __forceinline__ void epi_32(v8f acc[2][4], const float* bias,
                                       unsigned short* h, int lane, int wv,
                                       bool do_silu) {
#pragma unroll
  for (int mt = 0; mt < 2; ++mt) {
#pragma unroll
    for (int j = 0; j < 4; ++j) {
      int n = (wv * 4 + j) * 16 + (lane & 15);
      float bb = bias[n];
      int mb = mt * 16 + (lane >> 4) * 8;
#pragma unroll
      for (int v = 0; v < 8; ++v) {
        float x = acc[mt][j][v] + bb;
        if (do_silu) x = silu_f(x);
        h[(mb + v) * 256 + n] = f2bf(x);
      }
    }
  }
}

// ---------------------------------------------------------------------------
// Weight repack: row-major f32 (K x N) -> per-lane WMMA B fragments (bf16).
// out[((kt*NT+nt)*32+lane)*16+j] = W[kt*32 + (lane<16?0:8)+j+(j<8?0:8)][nt*16+lane%16]
// ---------------------------------------------------------------------------
__global__ void k_repack(const float* __restrict__ W, unsigned short* __restrict__ out,
                         int K, int N, int NT) {
  int idx = blockIdx.x * blockDim.x + threadIdx.x;
  int total = (K / 32) * NT * 32 * 16;
  if (idx >= total) return;
  int j    = idx & 15;
  int lane = (idx >> 4) & 31;
  int fi   = idx >> 9;
  int nt   = fi % NT;
  int kt   = fi / NT;
  int k = kt * 32 + ((lane < 16) ? 0 : 8) + j + ((j < 8) ? 0 : 8);
  int n = nt * 16 + (lane & 15);
  out[idx] = (n < N) ? f2bf(W[(size_t)k * N + n]) : (unsigned short)0;
}

__global__ void k_copy_f32(const float* __restrict__ src, float* __restrict__ dst, int n) {
  int i = blockIdx.x * blockDim.x + threadIdx.x;
  if (i < n) dst[i] = src[i];
}

// ---------------------------------------------------------------------------
// Time embedding: gfp(t) -> 256 -> 1024 (SiLU) -> 256.  One block per batch.
// ---------------------------------------------------------------------------
__global__ void k_time_embed(const float* __restrict__ t, const float* __restrict__ timeW,
                             const float* __restrict__ W1, const float* __restrict__ b1,
                             const float* __restrict__ W2, const float* __restrict__ b2,
                             float* __restrict__ temb) {
  __shared__ float g[256];
  __shared__ float h[1024];
  int b = blockIdx.x, tid = threadIdx.x;
  float tv = t[b];
  float xp = tv * timeW[tid & 127] * 6.283185307179586f;
  g[tid] = (tid < 128) ? __sinf(xp) : __cosf(xp);
  __syncthreads();
  for (int r = 0; r < 4; ++r) {
    int j = tid + r * 256;
    float acc = b1[j];
    for (int k = 0; k < 256; ++k) acc += g[k] * W1[(size_t)k * 1024 + j];
    h[j] = silu_f(acc);
  }
  __syncthreads();
  float acc = b2[tid];
  for (int k = 0; k < 1024; ++k) acc += h[k] * W2[(size_t)k * 256 + tid];
  temb[(size_t)b * 256 + tid] = acc;
}

// node = atom_embed[type] + t_emb[batch]  (bf16)
__global__ void k_node_init(const int* __restrict__ types, const int* __restrict__ batch,
                            const float* __restrict__ embed, const float* __restrict__ temb,
                            unsigned short* __restrict__ node_bf) {
  int idx = blockIdx.x * blockDim.x + threadIdx.x;
  if (idx >= NNODES * 256) return;
  int n = idx >> 8, f = idx & 255;
  float v = embed[(size_t)types[n] * 256 + f] + temb[(size_t)batch[n] * 256 + f];
  node_bf[idx] = f2bf(v);
}

// edge_attr = mlp2(edge_l1, edge_l2, gfp(|x_i-x_j|))  from INITIAL coords.
__global__ void k_edge_attr(const int* __restrict__ eidx, const float* __restrict__ coords0,
                            const float* __restrict__ edgeW,
                            const float* __restrict__ W1, const float* __restrict__ b1,
                            const float* __restrict__ W2, const float* __restrict__ b2,
                            unsigned short* __restrict__ eattr) {
  __shared__ float g[64];
  __shared__ float h[64];
  int e = blockIdx.x, tid = threadIdx.x;
  int r = eidx[e], c = eidx[EEDGES + e];
  float dx = coords0[r * 3 + 0] - coords0[c * 3 + 0];
  float dy = coords0[r * 3 + 1] - coords0[c * 3 + 1];
  float dz = coords0[r * 3 + 2] - coords0[c * 3 + 2];
  float d = sqrtf(dx * dx + dy * dy + dz * dz);
  float xp = d * edgeW[tid & 31] * 6.283185307179586f;
  g[tid] = (tid < 32) ? __sinf(xp) : __cosf(xp);
  __syncthreads();
  float acc = b1[tid];
  for (int k = 0; k < 64; ++k) acc += g[k] * W1[k * 64 + tid];
  h[tid] = silu_f(acc);
  __syncthreads();
  acc = b2[tid];
  for (int k = 0; k < 64; ++k) acc += h[k] * W2[k * 64 + tid];
  eattr[(size_t)e * 64 + tid] = f2bf(acc);
}

// CSR boundaries for the row-sorted edge list (no atomics needed later).
__global__ void k_row_start(const int* __restrict__ eidx, int* __restrict__ row_start) {
  int e = blockIdx.x * blockDim.x + threadIdx.x;
  if (e > EEDGES) return;
  if (e == 0) {
    int r0 = eidx[0];
    for (int r = 0; r <= r0; ++r) row_start[r] = 0;
  } else if (e == EEDGES) {
    int rl = eidx[EEDGES - 1];
    for (int r = rl + 1; r <= NNODES; ++r) row_start[r] = EEDGES;
  } else {
    int rp = eidx[e - 1], rc = eidx[e];
    for (int r = rp + 1; r <= rc; ++r) row_start[r] = e;
  }
}

// ---------------------------------------------------------------------------
// Edge MLP + fused coord-gate, 32 edges per workgroup (4 waves).
//   ef[32x576] -> WMMA(576->256)+SiLU -> WMMA(256->256)=em -> global bf16
//   em -> WMMA c1(256->256)+SiLU -> scalar c2 dot -> cm = s*cd/|cd|
// LDS overlay (ushort elems): ef [0,18432) | h1 [18432,26624)
//                             em [0,8192)  | h3 [8192,16384)   (ef dead)
// ---------------------------------------------------------------------------
__global__ void __launch_bounds__(128)
k_edge_mlp(const int* __restrict__ eidx,
           const unsigned short* __restrict__ node_bf,
           const unsigned short* __restrict__ eattr,
           const float* __restrict__ coords,
           const unsigned short* __restrict__ e1p, const float* __restrict__ e1b,
           const unsigned short* __restrict__ e2p, const float* __restrict__ e2b,
           const unsigned short* __restrict__ c1p, const float* __restrict__ c1b,
           const float* __restrict__ c2w, const float* __restrict__ c2b,
           unsigned short* __restrict__ embuf, float* __restrict__ cmbuf) {
  __shared__ unsigned short pool[26624];   // 52 KB
  __shared__ int s_row[32], s_col[32];
  unsigned short* s_ef = pool;
  unsigned short* s_h1 = pool + 18432;
  unsigned short* s_em = pool;             // overlays dead ef
  unsigned short* s_h3 = pool + 8192;      // overlays dead ef

  int tid = threadIdx.x;
  int lane = tid & 31;
  int wv = tid >> 5;
  int e0 = blockIdx.x * 32;

  if (tid < 32) {
    s_row[tid] = eidx[e0 + tid];
    s_col[tid] = eidx[EEDGES + e0 + tid];
  }
  __syncthreads();

  // vectorized gather: 32 rows x 72 uint4 (576 bf16 per row)
  for (int idx = tid; idx < 32 * 72; idx += 128) {
    int m = idx / 72, q = idx - m * 72;
    const uint4* src;
    if (q < 32)      src = (const uint4*)(node_bf + (size_t)s_row[m] * 256) + q;
    else if (q < 64) src = (const uint4*)(node_bf + (size_t)s_col[m] * 256) + (q - 32);
    else             src = (const uint4*)(eattr + (size_t)(e0 + m) * 64) + (q - 64);
    *((uint4*)(s_ef + m * 576) + q) = *src;
  }
  __syncthreads();

  {  // GEMM1: K=576 (18 k-tiles), SiLU
    v8f acc[2][4] = {};
    gemm_32xN(s_ef, 576, e1p, 18, lane, wv, acc);
    epi_32(acc, e1b, s_h1, lane, wv, true);
  }
  __syncthreads();

  {  // GEMM2: K=256 -> em (LDS)
    v8f acc[2][4] = {};
    gemm_32xN(s_h1, 256, e2p, 8, lane, wv, acc);
    epi_32(acc, e2b, s_em, lane, wv, false);
  }
  __syncthreads();

  // em LDS -> global (coalesced b128), then c-MLP on em (both read-only on s_em)
  for (int idx = tid; idx < 32 * 32; idx += 128)
    ((uint4*)(embuf + (size_t)e0 * 256))[idx] = ((const uint4*)s_em)[idx];

  {  // GEMM3: c1, SiLU
    v8f acc[2][4] = {};
    gemm_32xN(s_em, 256, c1p, 8, lane, wv, acc);
    epi_32(acc, c1b, s_h3, lane, wv, true);
  }
  __syncthreads();

  if (tid < 32) {  // c2 dot + coordinate gate
    int m = tid;
    float s = c2b[0];
    for (int k = 0; k < 256; ++k) s += bf2f(s_h3[m * 256 + k]) * c2w[k];
    int r = s_row[m], c = s_col[m];
    float dx = coords[r * 3 + 0] - coords[c * 3 + 0];
    float dy = coords[r * 3 + 1] - coords[c * 3 + 1];
    float dz = coords[r * 3 + 2] - coords[c * 3 + 2];
    float dist = sqrtf(dx * dx + dy * dy + dz * dz) + 1e-8f;
    float sc = s / dist;
    cmbuf[(size_t)(e0 + m) * 3 + 0] = sc * dx;
    cmbuf[(size_t)(e0 + m) * 3 + 1] = sc * dy;
    cmbuf[(size_t)(e0 + m) * 3 + 2] = sc * dz;
  }
}

// CSR segment-sum of em (->nm, bf16) and cm (->coords).
// One wave per node; each lane accumulates 8 features via b128 loads.
__global__ void k_node_reduce(const int* __restrict__ row_start,
                              const unsigned short* __restrict__ embuf,
                              const float* __restrict__ cmbuf,
                              unsigned short* __restrict__ nmbuf,
                              float* __restrict__ coords) {
  int n = blockIdx.x, t = threadIdx.x;   // 32 threads
  int s = row_start[n], e1 = row_start[n + 1];
  float acc[8] = {0.f, 0.f, 0.f, 0.f, 0.f, 0.f, 0.f, 0.f};
  for (int e = s; e < e1; ++e) {
    U4 u;
    u.q = ((const uint4*)(embuf + (size_t)e * 256))[t];
#pragma unroll
    for (int k = 0; k < 8; ++k) acc[k] += bf2f(u.u[k]);
  }
  U4 o;
#pragma unroll
  for (int k = 0; k < 8; ++k) o.u[k] = f2bf(acc[k]);
  ((uint4*)(nmbuf + (size_t)n * 256))[t] = o.q;
  if (t < 3) {
    float a = 0.f;
    for (int e = s; e < e1; ++e) a += cmbuf[(size_t)e * 3 + t];
    coords[n * 3 + t] += a;
  }
}

// node = mlp2(n1, n2, [node | nm]) ; 32 nodes per workgroup.
__global__ void __launch_bounds__(128)
k_node_mlp(const unsigned short* __restrict__ nmbuf,
           const unsigned short* __restrict__ n1p, const float* __restrict__ n1b,
           const unsigned short* __restrict__ n2p, const float* __restrict__ n2b,
           unsigned short* node_bf) {
  __shared__ unsigned short s_a[32 * 512];  // concat in; reused as out tile
  __shared__ unsigned short s_h[32 * 256];
  int tid = threadIdx.x, lane = tid & 31, wv = tid >> 5;
  int n0 = blockIdx.x * 32;

  for (int idx = tid; idx < 32 * 64; idx += 128) {   // 32 rows x 64 uint4
    int m = idx >> 6, q = idx & 63;
    const uint4* src = (q < 32)
        ? (const uint4*)(node_bf + (size_t)(n0 + m) * 256) + q
        : (const uint4*)(nmbuf + (size_t)(n0 + m) * 256) + (q - 32);
    ((uint4*)(s_a + m * 512))[q] = *src;
  }
  __syncthreads();

  {  // K=512, SiLU
    v8f acc[2][4] = {};
    gemm_32xN(s_a, 512, n1p, 16, lane, wv, acc);
    epi_32(acc, n1b, s_h, lane, wv, true);
  }
  __syncthreads();

  {  // K=256 -> result tile into s_a (32x256, stride 256)
    v8f acc[2][4] = {};
    gemm_32xN(s_h, 256, n2p, 8, lane, wv, acc);
    epi_32(acc, n2b, s_a, lane, wv, false);
  }
  __syncthreads();

  for (int idx = tid; idx < 32 * 32; idx += 128)
    ((uint4*)(node_bf + (size_t)n0 * 256))[idx] = ((const uint4*)s_a)[idx];
}

// Output heads: coord_noise (N,3) then type_logits (N,100), both f32.
__global__ void __launch_bounds__(128)
k_heads(const unsigned short* __restrict__ node_bf,
        const unsigned short* __restrict__ co1p, const float* __restrict__ co1b,
        const float* __restrict__ co2w, const float* __restrict__ co2b,
        const unsigned short* __restrict__ ty1p, const float* __restrict__ ty1b,
        const unsigned short* __restrict__ ty2p, const float* __restrict__ ty2b,
        float* __restrict__ out) {
  __shared__ unsigned short s_a[16 * 256];
  __shared__ unsigned short s_h[16 * 256];
  int tid = threadIdx.x, lane = tid & 31, wv = tid >> 5;
  int n0 = blockIdx.x * 16;

  for (int idx = tid; idx < 16 * 32; idx += 128)
    ((uint4*)s_a)[idx] = ((const uint4*)(node_bf + (size_t)n0 * 256))[idx];
  __syncthreads();

  {  // coord head hidden (M=16 variant: single M-tile)
    v8f acc[4] = {};
    for (int kt = 0; kt < 8; ++kt) {
      v16bf a = load_a_lds(s_a, 256, kt, lane);
#pragma unroll
      for (int j = 0; j < 4; ++j) {
        v16bf b = load_b_pk(co1p, kt, wv * 4 + j, 16, lane);
        acc[j] = wmma_bf16(a, b, acc[j]);
      }
    }
#pragma unroll
    for (int j = 0; j < 4; ++j) {
      int n = (wv * 4 + j) * 16 + (lane & 15);
      float bb = co1b[n];
      int mb = (lane >> 4) * 8;
#pragma unroll
      for (int v = 0; v < 8; ++v)
        s_h[(mb + v) * 256 + n] = f2bf(silu_f(acc[j][v] + bb));
    }
  }
  __syncthreads();
  if (tid < 48) {  // 256 -> 3, scalar
    int m = tid / 3, d = tid - m * 3;
    float acc = co2b[d];
    for (int k = 0; k < 256; ++k) acc += bf2f(s_h[m * 256 + k]) * co2w[k * 3 + d];
    out[(size_t)(n0 + m) * 3 + d] = acc;
  }
  __syncthreads();

  {  // type head hidden
    v8f acc[4] = {};
    for (int kt = 0; kt < 8; ++kt) {
      v16bf a = load_a_lds(s_a, 256, kt, lane);
#pragma unroll
      for (int j = 0; j < 4; ++j) {
        v16bf b = load_b_pk(ty1p, kt, wv * 4 + j, 16, lane);
        acc[j] = wmma_bf16(a, b, acc[j]);
      }
    }
#pragma unroll
    for (int j = 0; j < 4; ++j) {
      int n = (wv * 4 + j) * 16 + (lane & 15);
      float bb = ty1b[n];
      int mb = (lane >> 4) * 8;
#pragma unroll
      for (int v = 0; v < 8; ++v)
        s_h[(mb + v) * 256 + n] = f2bf(silu_f(acc[j][v] + bb));
    }
  }
  __syncthreads();

  float* logits = out + (size_t)NNODES * 3;
  for (int nt = wv; nt < 7; nt += 4) {  // N padded 100 -> 112 (7 tiles)
    v8f acc = {};
    for (int kt = 0; kt < 8; ++kt) {
      v16bf a = load_a_lds(s_h, 256, kt, lane);
      v16bf b = load_b_pk(ty2p, kt, nt, 7, lane);
      acc = wmma_bf16(a, b, acc);
    }
    int n = nt * 16 + (lane & 15);
    if (n < NSPEC) {
      float bb = ty2b[n];
      int mb = (lane >> 4) * 8;
#pragma unroll
      for (int v = 0; v < 8; ++v)
        logits[(size_t)(n0 + mb + v) * NSPEC + n] = acc[v] + bb;
    }
  }
}

// ---------------------------------------------------------------------------
extern "C" void kernel_launch(void* const* d_in, const int* in_sizes, int n_in,
                              void* d_out, int out_size, void* d_ws, size_t ws_size,
                              hipStream_t stream) {
  (void)in_sizes; (void)n_in; (void)out_size; (void)ws_size;

  // ---- input pointers (insertion-order flatten) ----
  const float* in_coords = (const float*)d_in[0];
  const float* in_t      = (const float*)d_in[1];
  const float* timeW     = (const float*)d_in[2];
  const float* tl1W      = (const float*)d_in[3];
  const float* tl1b      = (const float*)d_in[4];
  const float* tl2W      = (const float*)d_in[5];
  const float* tl2b      = (const float*)d_in[6];
  const float* atom_emb  = (const float*)d_in[7];
  const float* edgeW     = (const float*)d_in[8];
  const float* el1W      = (const float*)d_in[9];
  const float* el1b      = (const float*)d_in[10];
  const float* el2W      = (const float*)d_in[11];
  const float* el2b      = (const float*)d_in[12];
  struct LW { const float *e1W,*e1b,*e2W,*e2b,*n1W,*n1b,*n2W,*n2b,*c1W,*c1b,*c2W,*c2b; } L[6];
  for (int l = 0; l < 6; ++l) {
    int b = 13 + l * 12;
    L[l].e1W = (const float*)d_in[b + 0];  L[l].e1b = (const float*)d_in[b + 1];
    L[l].e2W = (const float*)d_in[b + 2];  L[l].e2b = (const float*)d_in[b + 3];
    L[l].n1W = (const float*)d_in[b + 4];  L[l].n1b = (const float*)d_in[b + 5];
    L[l].n2W = (const float*)d_in[b + 6];  L[l].n2b = (const float*)d_in[b + 7];
    L[l].c1W = (const float*)d_in[b + 8];  L[l].c1b = (const float*)d_in[b + 9];
    L[l].c2W = (const float*)d_in[b + 10]; L[l].c2b = (const float*)d_in[b + 11];
  }
  const float* co1W = (const float*)d_in[85];
  const float* co1b = (const float*)d_in[86];
  const float* co2W = (const float*)d_in[87];
  const float* co2b = (const float*)d_in[88];
  const float* ty1W = (const float*)d_in[89];
  const float* ty1b = (const float*)d_in[90];
  const float* ty2W = (const float*)d_in[91];
  const float* ty2b = (const float*)d_in[92];
  const int* atom_types = (const int*)d_in[93];
  const int* batch      = (const int*)d_in[94];
  const int* eidx       = (const int*)d_in[95];
  float* out = (float*)d_out;

  // ---- workspace carve (total ~49 MB) ----
  char* ws = (char*)d_ws;
  size_t off = 0;
  auto carve = [&](size_t bytes) {
    void* p = ws + off;
    off = (off + bytes + 255) & ~(size_t)255;
    return p;
  };
  float*          coords_ws = (float*)carve((size_t)NNODES * 3 * 4);
  float*          temb      = (float*)carve((size_t)NBATCH * 256 * 4);
  unsigned short* node_bf   = (unsigned short*)carve((size_t)NNODES * 256 * 2);
  unsigned short* nmbuf     = (unsigned short*)carve((size_t)NNODES * 256 * 2);
  unsigned short* eattr     = (unsigned short*)carve((size_t)EEDGES * 64 * 2);
  unsigned short* embuf     = (unsigned short*)carve((size_t)EEDGES * 256 * 2);
  float*          cmbuf     = (float*)carve((size_t)EEDGES * 3 * 4);
  int*            row_start = (int*)carve((size_t)(NNODES + 1) * 4);
  unsigned short* e1p[6]; unsigned short* e2p[6]; unsigned short* n1p[6];
  unsigned short* n2p[6]; unsigned short* c1p[6];
  for (int l = 0; l < 6; ++l) {
    e1p[l] = (unsigned short*)carve((size_t)576 * 256 * 2);
    e2p[l] = (unsigned short*)carve((size_t)256 * 256 * 2);
    n1p[l] = (unsigned short*)carve((size_t)512 * 256 * 2);
    n2p[l] = (unsigned short*)carve((size_t)256 * 256 * 2);
    c1p[l] = (unsigned short*)carve((size_t)256 * 256 * 2);
  }
  unsigned short* co1p = (unsigned short*)carve((size_t)256 * 256 * 2);
  unsigned short* ty1p = (unsigned short*)carve((size_t)256 * 256 * 2);
  unsigned short* ty2p = (unsigned short*)carve((size_t)256 * 112 * 2);

  auto repack = [&](const float* W, unsigned short* dst, int K, int N, int NT) {
    int total = (K / 32) * NT * 512;
    k_repack<<<(total + 255) / 256, 256, 0, stream>>>(W, dst, K, N, NT);
  };
  for (int l = 0; l < 6; ++l) {
    repack(L[l].e1W, e1p[l], 576, 256, 16);
    repack(L[l].e2W, e2p[l], 256, 256, 16);
    repack(L[l].n1W, n1p[l], 512, 256, 16);
    repack(L[l].n2W, n2p[l], 256, 256, 16);
    repack(L[l].c1W, c1p[l], 256, 256, 16);
  }
  repack(co1W, co1p, 256, 256, 16);
  repack(ty1W, ty1p, 256, 256, 16);
  repack(ty2W, ty2p, 256, 100, 7);

  k_copy_f32<<<(NNODES * 3 + 255) / 256, 256, 0, stream>>>(in_coords, coords_ws, NNODES * 3);
  k_time_embed<<<NBATCH, 256, 0, stream>>>(in_t, timeW, tl1W, tl1b, tl2W, tl2b, temb);
  k_node_init<<<(NNODES * 256) / 256, 256, 0, stream>>>(atom_types, batch, atom_emb, temb, node_bf);
  k_edge_attr<<<EEDGES, 64, 0, stream>>>(eidx, in_coords, edgeW, el1W, el1b, el2W, el2b, eattr);
  k_row_start<<<(EEDGES + 1 + 255) / 256, 256, 0, stream>>>(eidx, row_start);

  for (int l = 0; l < 6; ++l) {
    k_edge_mlp<<<EEDGES / 32, 128, 0, stream>>>(
        eidx, node_bf, eattr, coords_ws,
        e1p[l], L[l].e1b, e2p[l], L[l].e2b, c1p[l], L[l].c1b, L[l].c2W, L[l].c2b,
        embuf, cmbuf);
    k_node_reduce<<<NNODES, 32, 0, stream>>>(row_start, embuf, cmbuf, nmbuf, coords_ws);
    k_node_mlp<<<NNODES / 32, 128, 0, stream>>>(nmbuf, n1p[l], L[l].n1b, n2p[l], L[l].n2b, node_bf);
  }

  k_heads<<<NNODES / 16, 128, 0, stream>>>(node_bf, co1p, co1b, co2W, co2b,
                                           ty1p, ty1b, ty2p, ty2b, out);
}
